// MolecularMPNN_24008867185218
// MI455X (gfx1250) — matrix operand order repaired and checked by reading, not compiled, not run on hardware
//
#include <hip/hip_runtime.h>
#include <cstdint>
#include <cstddef>

// ---------------------------------------------------------------------------
// MolecularMPNN forward for gfx1250 (MI455X), wave32 + WMMA f16->f32 + TDM.
//
// Pipeline:
//   K0: pack W_e2 (f32 [64,1024]) -> W2Tg f16, pre-transposed + padded
//       layout [16 i][64 h][72 k]  (byte-identical to the LDS tile)
//   K1: eh = relu(edge_attr @ W_e1 + b_e1)                -> f16 [E,64]
//   K2: fused NNConv message:  per 16-edge tile,
//       for i in 0..15:  TDM  TENSOR_LOAD_TO_LDS (9216 B slice of W2Tg)
//                        We_i[16,64] = WMMA(eh_tile, slice) + b_e2
//                        msg += x[src][i] * We_i          (VALU, co-executed)
//       atomicAdd msg -> agg[dst], atomicAdd 1 -> cnt[dst]
//   K3: h_node = relu(agg/max(cnt,1) + x @ root + conv_bias)
//   K3b: gate = h_node @ Wg + bg
//   K4: per-graph attention softmax (batch is sorted) -> gctx [G,64]
//   K5: fused head: fc1 -> relu -> log_softmax head + sigmoid stop head
// ---------------------------------------------------------------------------

typedef __attribute__((ext_vector_type(16))) _Float16     v16h;
typedef __attribute__((ext_vector_type(8)))  float        v8f;
typedef __attribute__((ext_vector_type(4)))  unsigned int v4u;
typedef __attribute__((ext_vector_type(8)))  int          v8i;
typedef __attribute__((ext_vector_type(4)))  int          v4i;

#define NI 16
#define H  64
#define EI 8
#define GF 32
#define C  40
#define LDSK 72   // padded row stride (halves) for transposed W_e2 slice

// bytes per i-slice of the packed weight tensor (== LDS tile size)
#define W2T_SLICE_BYTES (H * LDSK * 2)      // 9216
#define W2T_SLICE_DW4   (W2T_SLICE_BYTES/4) // 2304 elements of data_size=4B

// ---------------------------------------------------------------- K0: pack W_e2 -> f16 transposed+padded
__global__ void k_pack_w2t(const float* __restrict__ W_e2,   // [64 k][1024 i*64+h]
                           _Float16* __restrict__ W2Tg)      // [16 i][64 h][72 k]
{
    int idx = blockIdx.x * blockDim.x + threadIdx.x;          // (i,h,k)
    if (idx >= NI * H * H) return;
    int k = idx & 63;
    int h = (idx >> 6) & 63;
    int i = idx >> 12;
    W2Tg[((size_t)i * H + h) * LDSK + k] =
        (_Float16)W_e2[(size_t)k * (NI * H) + i * H + h];
}

// ---------------------------------------------------------------- K1: edge MLP layer 1
__global__ void k_edge_mlp1(const float* __restrict__ edge_attr,
                            const float* __restrict__ W_e1,
                            const float* __restrict__ b_e1,
                            _Float16* __restrict__ eh, int E)
{
    long idx = (long)blockIdx.x * blockDim.x + threadIdx.x;   // one (e,h) pair
    if (idx >= (long)E * H) return;
    int e = (int)(idx >> 6), h = (int)(idx & 63);
    float acc = b_e1[h];
    const float* ar = edge_attr + (size_t)e * EI;
#pragma unroll
    for (int i = 0; i < EI; ++i) acc += ar[i] * W_e1[i * H + h];
    eh[idx] = (_Float16)fmaxf(acc, 0.f);
}

// ---------------------------------------------------------------- K2: fused NNConv message (TDM + WMMA)
__global__ __launch_bounds__(256)
void k_nnconv_msg(const _Float16* __restrict__ eh,      // [E,64]
                  const _Float16* __restrict__ W2Tg,    // [16][64][72] f16 packed
                  const float* __restrict__ b_e2,       // [1024]
                  const float* __restrict__ x,          // [N,16]
                  const int* __restrict__ srcArr,
                  const int* __restrict__ dstArr,
                  float* __restrict__ agg,              // [N,64]
                  float* __restrict__ cnt,              // [N]
                  int E)
{
    __shared__ _Float16 w2t[H][LDSK];     // one i-slice: [h_local][k], 9216 B
    __shared__ float    xs[8][16][17];    // per-wave x[src] tiles: [wave][edge][i]

    const int tid     = threadIdx.x;
    const int wave    = tid >> 5;
    const int lane    = tid & 31;
    const int half_id = lane >> 4;        // 0: lanes 0-15, 1: lanes 16-31
    const int nlane   = lane & 15;
    const int e0      = (blockIdx.x * 8 + wave) * 16;   // this wave's 16-edge tile
    const bool active = (e0 < E);                       // E % 16 == 0 -> full tiles

    // ---- stage this wave's x[src] tile (wave-private, no barrier needed)
    if (active) {
        int e = e0 + nlane;
        int s = srcArr[e];
        const float* xr = x + (size_t)s * NI + half_id * 8;
#pragma unroll
        for (int c = 0; c < 8; ++c) xs[wave][nlane][half_id * 8 + c] = xr[c];
    }

    // ---- A fragments: eh rows, K = 0..31 (a0) and 32..63 (a1)
    // 16-bit A 16x32 striping: lane m=lane%16; element j -> k = j + (j<8?0:8) + half_id*8
    v16h a0 = {}, a1 = {};
    if (active) {
        const _Float16* rowp = eh + (size_t)(e0 + nlane) * H;
#pragma unroll
        for (int j = 0; j < 8; ++j) {
            a0[j]     = rowp[half_id * 8 + j];
            a0[j + 8] = rowp[16 + half_id * 8 + j];
            a1[j]     = rowp[32 + half_id * 8 + j];
            a1[j + 8] = rowp[48 + half_id * 8 + j];
        }
    }

    // msg accumulators: [16 edges][64 h] as 4 C-tiles of v8f
    v8f msg[4];
#pragma unroll
    for (int hb = 0; hb < 4; ++hb)
#pragma unroll
        for (int r = 0; r < 8; ++r) msg[hb][r] = 0.f;

    const unsigned lds_base = (unsigned)(size_t)(&w2t[0][0]);

    for (int i = 0; i < NI; ++i) {
        __syncthreads();   // previous slice fully consumed by all waves
        if (wave == 0) {
            // ---- TDM: one contiguous 9216-B copy global -> LDS (D# per ISA ch.8)
            unsigned long long ga =
                (unsigned long long)(size_t)(W2Tg + (size_t)i * (H * LDSK));
            v4u g0;
            g0[0] = __builtin_amdgcn_readfirstlane(1u);               // count=1, user D#
            g0[1] = __builtin_amdgcn_readfirstlane(lds_base);         // lds_addr
            g0[2] = __builtin_amdgcn_readfirstlane((unsigned)(ga & 0xFFFFFFFFu));
            g0[3] = __builtin_amdgcn_readfirstlane(
                        ((unsigned)((ga >> 32) & 0x1FFFFFFu)) | (2u << 30)); // type=2
            v8i g1;
            g1[0] = (int)(2u << 16);             // data_size = 4 B, no multicast
            g1[1] = (int)(W2T_SLICE_DW4 << 16);  // tensor_dim0 (low 16 -> bits 63:48)
            g1[2] = (int)(1u << 16);             // tensor_dim1 = 1
            g1[3] = (int)(W2T_SLICE_DW4 << 16);  // tile_dim0 (bits 127:112)
            g1[4] = 0;                           // tile_dim1/2 unused
            g1[5] = (int)W2T_SLICE_DW4;          // tensor_dim0_stride
            g1[6] = 0;
            g1[7] = 0;
            v4i gz4 = {0, 0, 0, 0};              // groups 2/3 unused (<=2D tensor)
            v8i gz8 = {0, 0, 0, 0, 0, 0, 0, 0};  // extra group (6-arg form), unused
            __builtin_amdgcn_tensor_load_to_lds(g0, g1, gz4, gz4, gz8, 0);
            __builtin_amdgcn_s_wait_tensorcnt(0);
        }
        __syncthreads();   // slice visible to every wave

        if (active) {
#pragma unroll
            for (int hb = 0; hb < 4; ++hb) {
                // C init = b_e2 broadcast (depends only on n = h index)
                float bias = b_e2[i * H + hb * 16 + nlane];
                v8f c;
#pragma unroll
                for (int r = 0; r < 8; ++r) c[r] = bias;

                // B fragments from LDS: element j -> k = kb*32 + j + (j<8?0:8) + half_id*8,
                // row = hb*16 + nlane. Two 8-half runs.
                const _Float16* wr = &w2t[hb * 16 + nlane][half_id * 8];
                v16h b0, b1;
#pragma unroll
                for (int j = 0; j < 16; ++j) { b0[j] = wr[j + (j < 8 ? 0 : 8)]; b1[j] = wr[32 + j + (j < 8 ? 0 : 8)]; }

                c = __builtin_amdgcn_wmma_f32_16x16x32_f16(
                        false, a0, false, b0, (short)0, c, false, false);
                c = __builtin_amdgcn_wmma_f32_16x16x32_f16(
                        false, a1, false, b1, (short)0, c, false, false);

                // msg[e][h] += x[src[e]][i] * We_i[e][h]   (row scalar per edge)
#pragma unroll
                for (int r = 0; r < 8; ++r)
                    msg[hb][r] += xs[wave][half_id * 8 + r][i] * c[r];
            }
        }
    }

    if (active) {
#pragma unroll
        for (int r = 0; r < 8; ++r) {
            int m = half_id * 8 + r;             // edge row within tile
            int d = dstArr[e0 + m];
#pragma unroll
            for (int hb = 0; hb < 4; ++hb)
                atomicAdd(&agg[(size_t)d * H + hb * 16 + nlane], msg[hb][r]);
        }
        if (half_id == 0) atomicAdd(&cnt[dstArr[e0 + nlane]], 1.0f);
    }
}

// ---------------------------------------------------------------- K3: node update
__global__ void k_node_update(const float* __restrict__ x,
                              const float* __restrict__ agg,
                              const float* __restrict__ cnt,
                              const float* __restrict__ root,
                              const float* __restrict__ conv_bias,
                              float* __restrict__ h_node, int N)
{
    long idx = (long)blockIdx.x * blockDim.x + threadIdx.x;   // (n,h)
    if (idx >= (long)N * H) return;
    int n = (int)(idx >> 6), h = (int)(idx & 63);
    float inv = 1.f / fmaxf(cnt[n], 1.f);
    float acc = agg[idx] * inv + conv_bias[h];
    const float* xr = x + (size_t)n * NI;
#pragma unroll
    for (int i = 0; i < NI; ++i) acc += xr[i] * root[i * H + h];
    h_node[idx] = fmaxf(acc, 0.f);
}

// ---------------------------------------------------------------- K3b: gate
__global__ void k_gate(const float* __restrict__ h_node,
                       const float* __restrict__ Wg,
                       const float* __restrict__ bg,
                       float* __restrict__ gate, int N)
{
    int n = blockIdx.x * blockDim.x + threadIdx.x;
    if (n >= N) return;
    float acc = bg[0];
    const float* hr = h_node + (size_t)n * H;
#pragma unroll
    for (int h = 0; h < H; ++h) acc += hr[h] * Wg[h];
    gate[n] = acc;
}

// ---------------------------------------------------------------- K4: per-graph attention pool
__global__ __launch_bounds__(64)
void k_attention(const float* __restrict__ gate,
                 const float* __restrict__ h_node,
                 const int* __restrict__ batch,
                 float* __restrict__ gctx, int N)
{
    __shared__ float red[64];
    const int g = blockIdx.x;
    const int tid = threadIdx.x;

    // lower_bound in sorted batch
    int lo = 0, hi = N;
    while (lo < hi) { int mid = (lo + hi) >> 1; if (batch[mid] < g) lo = mid + 1; else hi = mid; }
    const int s = lo;
    lo = s; hi = N;
    while (lo < hi) { int mid = (lo + hi) >> 1; if (batch[mid] < g + 1) lo = mid + 1; else hi = mid; }
    const int e = lo;

    // segment max
    float m = -3.0e38f;
    for (int n = s + tid; n < e; n += 64) m = fmaxf(m, gate[n]);
    red[tid] = m; __syncthreads();
    for (int off = 32; off > 0; off >>= 1) {
        if (tid < off) red[tid] = fmaxf(red[tid], red[tid + off]);
        __syncthreads();
    }
    m = red[0]; __syncthreads();

    // segment sum of exp
    float sum = 0.f;
    for (int n = s + tid; n < e; n += 64) sum += __expf(gate[n] - m);
    red[tid] = sum; __syncthreads();
    for (int off = 32; off > 0; off >>= 1) {
        if (tid < off) red[tid] += red[tid + off];
        __syncthreads();
    }
    float denom = red[0];

    // gctx[g][tid] = sum_n attn * h_node[n][tid]   (tid = channel, coalesced)
    float acc = 0.f;
    if (e > s && denom > 0.f) {
        for (int n = s; n < e; ++n)
            acc += __expf(gate[n] - m) * h_node[(size_t)n * H + tid];
        acc /= denom;
    }
    gctx[(size_t)g * H + tid] = acc;
}

// ---------------------------------------------------------------- K5: fused head
__global__ void k_final(const float* __restrict__ h_node,
                        const float* __restrict__ gctx,
                        const int* __restrict__ batch,
                        const float* __restrict__ pocket,
                        const float* __restrict__ W_fc1,  // [160,64]
                        const float* __restrict__ b_fc1,
                        const float* __restrict__ W_np,   // [64,40]
                        const float* __restrict__ b_np,
                        const float* __restrict__ W_sp,   // [64]
                        const float* __restrict__ b_sp,
                        float* __restrict__ out_pred,     // [N,40]
                        float* __restrict__ out_stop,     // [N]
                        int N)
{
    int n = blockIdx.x * blockDim.x + threadIdx.x;
    if (n >= N) return;
    int g = batch[n];

    float f[H];
#pragma unroll
    for (int h = 0; h < H; ++h) f[h] = b_fc1[h];
    float sgate = b_sp[0];

    const float* hr = h_node + (size_t)n * H;
    for (int j = 0; j < H; ++j) {
        float v = hr[j];
#pragma unroll
        for (int h = 0; h < H; ++h) f[h] += v * W_fc1[j * H + h];
    }
    const float* gr = gctx + (size_t)g * H;
    for (int j = 0; j < H; ++j) {
        float v = gr[j];
        sgate += v * W_sp[j];
#pragma unroll
        for (int h = 0; h < H; ++h) f[h] += v * W_fc1[(H + j) * H + h];
    }
    const float* pr = pocket + (size_t)g * GF;
    for (int j = 0; j < GF; ++j) {
        float v = pr[j];
#pragma unroll
        for (int h = 0; h < H; ++h) f[h] += v * W_fc1[(2 * H + j) * H + h];
    }
#pragma unroll
    for (int h = 0; h < H; ++h) f[h] = fmaxf(f[h], 0.f);

    float logits[C];
    float maxl = -3.0e38f;
    for (int c = 0; c < C; ++c) {
        float acc = b_np[c];
#pragma unroll
        for (int h = 0; h < H; ++h) acc += f[h] * W_np[h * C + c];
        logits[c] = acc;
        maxl = fmaxf(maxl, acc);
    }
    float lse = 0.f;
#pragma unroll
    for (int c = 0; c < C; ++c) lse += __expf(logits[c] - maxl);
    lse = __logf(lse);
    float* op = out_pred + (size_t)n * C;
#pragma unroll
    for (int c = 0; c < C; ++c) op[c] = logits[c] - maxl - lse;

    out_stop[n] = 1.f / (1.f + __expf(-sgate));
}

// ---------------------------------------------------------------- launcher
extern "C" void kernel_launch(void* const* d_in, const int* in_sizes, int n_in,
                              void* d_out, int out_size, void* d_ws, size_t ws_size,
                              hipStream_t stream)
{
    (void)n_in; (void)out_size; (void)ws_size;

    const float* x          = (const float*)d_in[0];
    const int*   edge_index = (const int*)  d_in[1];
    const float* edge_attr  = (const float*)d_in[2];
    const int*   batch      = (const int*)  d_in[3];
    const float* pocket     = (const float*)d_in[4];
    const float* W_e1 = (const float*)d_in[5];  const float* b_e1 = (const float*)d_in[6];
    const float* W_e2 = (const float*)d_in[7];  const float* b_e2 = (const float*)d_in[8];
    const float* root = (const float*)d_in[9];  const float* conv_bias = (const float*)d_in[10];
    const float* Wg   = (const float*)d_in[11]; const float* bg   = (const float*)d_in[12];
    const float* W_fc1= (const float*)d_in[13]; const float* b_fc1= (const float*)d_in[14];
    const float* W_np = (const float*)d_in[15]; const float* b_np = (const float*)d_in[16];
    const float* W_sp = (const float*)d_in[17]; const float* b_sp = (const float*)d_in[18];

    const int N = in_sizes[0] / NI;
    const int E = in_sizes[1] / 2;
    const int G = in_sizes[4] / GF;

    const int* srcArr = edge_index;
    const int* dstArr = edge_index + E;

    // workspace carve-up (base is 256-aligned; ~156 MB total)
    char* base = (char*)d_ws;
    size_t off = 0;
    auto carve = [&](size_t bytes) { char* p = base + off; off += (bytes + 255) & ~(size_t)255; return p; };

    _Float16* eh   = (_Float16*)carve((size_t)E * H * sizeof(_Float16));
    _Float16* W2Tg = (_Float16*)carve((size_t)NI * H * LDSK * sizeof(_Float16));
    float* agg     = (float*)carve((size_t)N * H * sizeof(float) + (size_t)N * sizeof(float)); // agg+cnt contiguous
    float* cnt     = agg + (size_t)N * H;
    float* h_node  = (float*)carve((size_t)N * H * sizeof(float));
    float* gate    = (float*)carve((size_t)N * sizeof(float));
    float* gctx    = (float*)carve((size_t)G * H * sizeof(float));

    float* out_pred = (float*)d_out;
    float* out_stop = out_pred + (size_t)N * C;

    // zero the scatter targets (agg + cnt, contiguous)
    (void)hipMemsetAsync(agg, 0, (size_t)N * H * sizeof(float) + (size_t)N * sizeof(float), stream);

    {   // K0: pack weights (tiny)
        int total = NI * H * H;
        k_pack_w2t<<<(total + 255) / 256, 256, 0, stream>>>(W_e2, W2Tg);
    }
    {   // K1
        long total = (long)E * H;
        int grid = (int)((total + 255) / 256);
        k_edge_mlp1<<<grid, 256, 0, stream>>>(edge_attr, W_e1, b_e1, eh, E);
    }
    {   // K2: 8 waves/block, 16 edges/wave -> 128 edges/block
        int grid = (E + 127) / 128;
        k_nnconv_msg<<<grid, 256, 0, stream>>>(eh, W2Tg, b_e2, x, srcArr, dstArr, agg, cnt, E);
    }
    {   // K3
        long total = (long)N * H;
        int grid = (int)((total + 255) / 256);
        k_node_update<<<grid, 256, 0, stream>>>(x, agg, cnt, root, conv_bias, h_node, N);
    }
    {   // K3b
        int grid = (N + 255) / 256;
        k_gate<<<grid, 256, 0, stream>>>(h_node, Wg, bg, gate, N);
    }
    {   // K4
        k_attention<<<G, 64, 0, stream>>>(gate, h_node, batch, gctx, N);
    }
    {   // K5
        int grid = (N + 255) / 256;
        k_final<<<grid, 256, 0, stream>>>(h_node, gctx, batch, pocket,
                                          W_fc1, b_fc1, W_np, b_np, W_sp, b_sp,
                                          out_pred, out_stop, N);
    }
}
